// SimpleMHA_6682969112680
// MI455X (gfx1250) — matrix operand order repaired and checked by reading, not compiled, not run on hardware
//
#include <hip/hip_runtime.h>
#include <math.h>

// ---------------- problem constants ----------------
constexpr int BB = 4;          // batch
constexpr int SS = 2048;       // seq len
constexpr int EE = 1024;       // embed
constexpr int HH = 16;         // heads
constexpr int DD = 64;         // head dim
constexpr float SCALE = 0.125f;   // 1/sqrt(64)

typedef __attribute__((ext_vector_type(16))) _Float16 v16h;
typedef __attribute__((ext_vector_type(8)))  _Float16 v8h;
typedef __attribute__((ext_vector_type(8)))  float    v8f;
typedef __attribute__((ext_vector_type(4)))  unsigned int v4u;
typedef __attribute__((ext_vector_type(8)))  int      v8i;
typedef __attribute__((ext_vector_type(4)))  int      v4i;

#define DEV __device__ __forceinline__

#if defined(__has_builtin)
#if __has_builtin(__builtin_amdgcn_tensor_load_to_lds)
#define HAS_TDM 1
#else
#define HAS_TDM 0
#endif
#else
#define HAS_TDM 0
#endif

union V16U { v16h v; v8h h[2]; };

DEV v8f wmma16(v16h a, v16h b, v8f c) {
    return __builtin_amdgcn_wmma_f32_16x16x32_f16(false, a, false, b, (short)0, c, false, false);
}

// A-matrix fragment (16x32, M x K), row-major source, tile base = (row0,k0).
// Lane (m = lane&15, half = lane>>4) holds 16 contiguous K at half*16.
DEV v16h load_a(const _Float16* A, int ld, int lane) {
    int m = lane & 15, half = (lane >> 4) & 1;
    return *(const v16h*)(A + (size_t)m * ld + half * 16);
}

// B-matrix fragment (32x16, K x N) loaded from an N x K row-major array (i.e. B^T),
// tile base = (n0, k0). Lane (n = lane&15, half) reads 16 contiguous K at half*16.
DEV v16h load_bt(const _Float16* Bt, int ld, int lane) {
    int n = lane & 15, half = (lane >> 4) & 1;
    return *(const v16h*)(Bt + (size_t)n * ld + half * 16);
}

DEV float rmax16(float x) {
    x = fmaxf(x, __shfl_xor(x, 1, 32));
    x = fmaxf(x, __shfl_xor(x, 2, 32));
    x = fmaxf(x, __shfl_xor(x, 4, 32));
    x = fmaxf(x, __shfl_xor(x, 8, 32));
    return x;
}
DEV float rsum16(float x) {
    x += __shfl_xor(x, 1, 32);
    x += __shfl_xor(x, 2, 32);
    x += __shfl_xor(x, 4, 32);
    x += __shfl_xor(x, 8, 32);
    return x;
}

#if HAS_TDM
// Issue a TDM 2D tile load (f16 elements) into LDS at byte offset lds_off.
// Descriptor packing per CDNA5 ISA 8.3/8.4: group0 {count=1 | lds_addr | global_addr | type=2},
// group1 {data_size=2B, tensor_dim0/1 = tile dims (tile is in-bounds), tile_dim0/1, dim0 stride}.
DEV void tdm_load_2d(unsigned lds_off, const void* g, unsigned tile0, unsigned tile1,
                     unsigned stride0) {
    unsigned long long ga = (unsigned long long)g;
    unsigned g_lo = __builtin_amdgcn_readfirstlane((unsigned)ga);
    unsigned g_hi = __builtin_amdgcn_readfirstlane((unsigned)(ga >> 32));
    lds_off = __builtin_amdgcn_readfirstlane(lds_off);
    v4u g0;
    g0[0] = 1u;                                        // count=1, load, user mode
    g0[1] = lds_off;                                   // lds_addr (bytes)
    g0[2] = g_lo;                                      // global_addr[31:0]
    g0[3] = (g_hi & 0x01FFFFFFu) | (2u << 30);         // global_addr[56:32] | type=2
    v8i g1;
    g1[0] = (int)(1u << 16);                           // data_size = 1 -> 2 bytes
    g1[1] = (int)((tile0 & 0xFFFFu) << 16);            // tensor_dim0[15:0] @ [63:48]
    g1[2] = (int)(((tile0 >> 16) & 0xFFFFu) | ((tile1 & 0xFFFFu) << 16)); // dim0 hi | dim1 lo
    g1[3] = (int)(((tile1 >> 16) & 0xFFFFu) | ((tile0 & 0xFFFFu) << 16)); // dim1 hi | tile_dim0
    g1[4] = (int)(tile1 & 0xFFFFu);                    // tile_dim1, tile_dim2 = 0
    g1[5] = (int)stride0;                              // tensor_dim0_stride[31:0]
    g1[6] = 0;
    g1[7] = 0;
    v4i z4 = {0, 0, 0, 0};
#if __clang_major__ >= 23
    v8i z8 = {0, 0, 0, 0, 0, 0, 0, 0};
    __builtin_amdgcn_tensor_load_to_lds(g0, g1, z4, z4, z8, 0);
#else
    __builtin_amdgcn_tensor_load_to_lds(g0, g1, z4, z4, 0);
#endif
}
#endif

// ---------------- helpers ----------------
__global__ void cvt_f32_f16(const float* __restrict__ src, _Float16* __restrict__ dst, int n) {
    int i = blockIdx.x * blockDim.x + threadIdx.x;
    int stride = gridDim.x * blockDim.x;
    for (; i < n; i += stride) dst[i] = (_Float16)src[i];
}

__global__ void zero_f32(float* __restrict__ p, size_t n) {
    size_t i = (size_t)blockIdx.x * blockDim.x + threadIdx.x;
    size_t stride = (size_t)gridDim.x * blockDim.x;
    for (; i < n; i += stride) p[i] = 0.f;
}

// ---------------- QKV projection ----------------
// y = x @ W^T + b ; per wave: one 16x64 tile (A-frag reused across 4 N-subtiles).
__global__ __launch_bounds__(256) void qkv_proj(
    const _Float16* __restrict__ xh,
    const _Float16* __restrict__ wq, const _Float16* __restrict__ wk, const _Float16* __restrict__ wv,
    const float* __restrict__ bq, const float* __restrict__ bk, const float* __restrict__ bv,
    _Float16* __restrict__ Q, _Float16* __restrict__ K, _Float16* __restrict__ Vt)
{
    const int MT = (BB * SS) / 16;   // 512
    const int NT4 = EE / 64;         // 16 groups of 4 n-tiles
    int wid = blockIdx.x * (blockDim.x >> 5) + (threadIdx.x >> 5);
    if (wid >= 3 * MT * NT4) return;
    int which = wid / (MT * NT4);
    int rem   = wid % (MT * NT4);
    int mt = rem / NT4, ng = rem % NT4;
    int lane = threadIdx.x & 31, n = lane & 15, half = (lane >> 4) & 1;

    const _Float16* W    = (which == 0) ? wq : (which == 1) ? wk : wv;
    const float*    bias = (which == 0) ? bq : (which == 1) ? bk : bv;

    const _Float16* Arow = xh + (size_t)(mt * 16) * EE;
    const _Float16* Bt   = W  + (size_t)(ng * 64) * EE;

    v8f acc[4] = {{}, {}, {}, {}};
    #pragma unroll 2
    for (int k0 = 0; k0 < EE; k0 += 32) {
        v16h a = load_a(Arow + k0, EE, lane);
        #pragma unroll
        for (int ns = 0; ns < 4; ++ns) {
            v16h b = load_bt(Bt + (size_t)(ns * 16) * EE + k0, EE, lane);
            acc[ns] = wmma16(a, b, acc[ns]);
        }
    }

    #pragma unroll
    for (int ns = 0; ns < 4; ++ns) {
        int f = ng * 64 + ns * 16 + n;
        int h = f >> 6, d = f & 63;
        float bi = bias[f];
        #pragma unroll
        for (int v = 0; v < 8; ++v) {
            int row = mt * 16 + v + 8 * half;       // global (b*S+s)
            int b   = row >> 11;
            int s   = row & (SS - 1);
            float val = acc[ns][v] + bi;
            if (which == 0)
                Q[(((size_t)b * HH + h) * SS + s) * DD + d] = (_Float16)val;
            else if (which == 1)
                K[(((size_t)b * HH + h) * SS + s) * DD + d] = (_Float16)val;
            else
                Vt[(((size_t)b * HH + h) * DD + d) * SS + s] = (_Float16)val;
        }
    }
}

// ---------------- attention (flash pass + attn_avg pass) ----------------
// one wave = one (b, h, 16-row query tile); K/V tiles double-buffered into LDS by the TDM.
// per-wave LDS (elements): K0,K1 (32x64 each) | V0,V1 (64x32 each) | P (16x32)
constexpr int WV_K   = 32 * 64;         // 2048
constexpr int WV_ELE = 4 * WV_K + 512;  // 8704 elements = 17408 bytes per wave
__global__ __launch_bounds__(128) void attn_kernel(
    const _Float16* __restrict__ Q, const _Float16* __restrict__ K, const _Float16* __restrict__ Vt,
    _Float16* __restrict__ ctx, float* __restrict__ attn)
{
    __shared__ alignas(128) _Float16 smem[4 * WV_ELE];

    int wslot = threadIdx.x >> 5;
    int wid = blockIdx.x * 4 + wslot;
    int it = wid & (SS / 16 - 1);               // 128 query tiles
    int bh = wid >> 7;
    int h = bh & (HH - 1), b = bh >> 4;
    int s0 = it * 16;
    int lane = threadIdx.x & 31, n = lane & 15, half = (lane >> 4) & 1;

    const _Float16* Qh = Q  + (size_t)(b * HH + h) * SS * DD;
    const _Float16* Kh = K  + (size_t)(b * HH + h) * SS * DD;
    const _Float16* Vh = Vt + (size_t)(b * HH + h) * DD * SS;

    _Float16* wbase = &smem[wslot * WV_ELE];
    _Float16* kb[2] = { wbase, wbase + WV_K };
    _Float16* vb[2] = { wbase + 2 * WV_K, wbase + 3 * WV_K };
    _Float16* lp    = wbase + 4 * WV_K;
    unsigned wboff  = (unsigned)(wslot * WV_ELE) * 2u;   // byte offset of wave region in LDS

    v16h aq0 = load_a(Qh + (size_t)s0 * DD,      DD, lane);
    v16h aq1 = load_a(Qh + (size_t)s0 * DD + 32, DD, lane);

    float m8[8], l8[8];
    v8f cacc[4] = {{}, {}, {}, {}};
    #pragma unroll
    for (int v = 0; v < 8; ++v) { m8[v] = -1e30f; l8[v] = 0.f; }

#if HAS_TDM
    // prologue: stage first K (32x64) and V (64x32) tiles
    tdm_load_2d(wboff,                 Kh, 64, 32, 64);
    tdm_load_2d(wboff + 4 * WV_K,      Vh, 32, 64, SS);
#endif

    // ---- pass 1: flash softmax + P@V ----
    for (int i = 0; i < SS / 32; ++i) {
        int j0 = i * 32;
        int buf = i & 1;
#if HAS_TDM
        if (i + 1 < SS / 32) {
            // all prior ds reads of the other buffer must retire before TDM overwrites it
            asm volatile("s_wait_dscnt 0" ::: "memory");
            unsigned nb = (unsigned)(buf ^ 1);
            tdm_load_2d(wboff + nb * (2 * WV_K),            Kh + (size_t)(j0 + 32) * DD, 64, 32, 64);
            tdm_load_2d(wboff + 4 * WV_K + nb * (2 * WV_K), Vh + (j0 + 32),              32, 64, SS);
            __builtin_amdgcn_s_wait_tensorcnt(2);   // current buffer's two loads are done
        } else {
            __builtin_amdgcn_s_wait_tensorcnt(0);
        }
        asm volatile("" ::: "memory");
        const _Float16* Kt0 = kb[buf];          // 32 rows x 64, ld=64
        const _Float16* Vt0 = vb[buf];          // 64 rows x 32, ld=32
        const int ldk = 64, ldv = 32;
#else
        const _Float16* Kt0 = Kh + (size_t)j0 * DD;   // global fallback
        const _Float16* Vt0 = Vh + j0;
        const int ldk = DD, ldv = SS;
#endif
        v8f c[2];
        #pragma unroll
        for (int t = 0; t < 2; ++t) {
            const _Float16* Kt = Kt0 + (size_t)(16 * t) * ldk;
            v16h b0 = load_bt(Kt,      ldk, lane);
            v16h b1 = load_bt(Kt + 32, ldk, lane);
            v8f s = {};
            s = wmma16(aq0, b0, s);
            s = wmma16(aq1, b1, s);
            c[t] = s * SCALE;
        }
        float efv[8];
        #pragma unroll
        for (int v = 0; v < 8; ++v) {
            float mx = rmax16(fmaxf(c[0][v], c[1][v]));
            float mn = fmaxf(m8[v], mx);
            float e0 = __expf(c[0][v] - mn);
            float e1 = __expf(c[1][v] - mn);
            c[0][v] = e0; c[1][v] = e1;
            float sum = rsum16(e0 + e1);
            float ef = __expf(m8[v] - mn);
            l8[v] = l8[v] * ef + sum;
            m8[v] = mn;
            efv[v] = ef;
        }
        #pragma unroll
        for (int dt = 0; dt < 4; ++dt)
            #pragma unroll
            for (int v = 0; v < 8; ++v) cacc[dt][v] *= efv[v];

        // P (C-layout) -> LDS -> A-layout
        #pragma unroll
        for (int t = 0; t < 2; ++t)
            #pragma unroll
            for (int v = 0; v < 8; ++v)
                lp[(v + 8 * half) * 32 + t * 16 + n] = (_Float16)c[t][v];
        asm volatile("s_wait_dscnt 0" ::: "memory");
        v16h ap = load_a(lp, 32, lane);

        #pragma unroll
        for (int dt = 0; dt < 4; ++dt) {
            v16h bv = load_bt(Vt0 + (size_t)(dt * 16) * ldv, ldv, lane);
            cacc[dt] = wmma16(ap, bv, cacc[dt]);
        }
    }

    float invl[8];
    #pragma unroll
    for (int v = 0; v < 8; ++v) invl[v] = 1.f / l8[v];

    // store ctx as [B,S,E] f16
    #pragma unroll
    for (int dt = 0; dt < 4; ++dt) {
        int f = h * 64 + dt * 16 + n;
        #pragma unroll
        for (int v = 0; v < 8; ++v) {
            int row = s0 + v + 8 * half;
            ctx[((size_t)b * SS + row) * EE + f] = (_Float16)(cacc[dt][v] * invl[v]);
        }
    }

    // ---- pass 2: attn_avg accumulation ----
    float wsc[8];
    #pragma unroll
    for (int v = 0; v < 8; ++v) wsc[v] = invl[v] * (1.0f / HH);

    for (int j0 = 0; j0 < SS; j0 += 16) {
        const _Float16* Kt = Kh + (size_t)j0 * DD;
        v16h b0 = load_bt(Kt,      DD, lane);
        v16h b1 = load_bt(Kt + 32, DD, lane);
        v8f s = {};
        s = wmma16(aq0, b0, s);
        s = wmma16(aq1, b1, s);
        s = s * SCALE;
        #pragma unroll
        for (int v = 0; v < 8; ++v) {
            int row = s0 + v + 8 * half;
            float p = __expf(s[v] - m8[v]) * wsc[v];
            atomicAdd(&attn[((size_t)b * SS + row) * SS + j0 + n], p);
        }
    }
}

// ---------------- output projection ----------------
__global__ __launch_bounds__(256) void out_proj(
    const _Float16* __restrict__ ctx, const _Float16* __restrict__ wo,
    const float* __restrict__ bo, float* __restrict__ out)
{
    const int MT = (BB * SS) / 16;   // 512
    const int NT4 = EE / 64;         // 16
    int wid = blockIdx.x * (blockDim.x >> 5) + (threadIdx.x >> 5);
    if (wid >= MT * NT4) return;
    int mt = wid / NT4, ng = wid % NT4;
    int lane = threadIdx.x & 31, n = lane & 15, half = (lane >> 4) & 1;

    const _Float16* Arow = ctx + (size_t)(mt * 16) * EE;
    const _Float16* Bt   = wo  + (size_t)(ng * 64) * EE;

    v8f acc[4] = {{}, {}, {}, {}};
    #pragma unroll 2
    for (int k0 = 0; k0 < EE; k0 += 32) {
        v16h a = load_a(Arow + k0, EE, lane);
        #pragma unroll
        for (int ns = 0; ns < 4; ++ns) {
            v16h b = load_bt(Bt + (size_t)(ns * 16) * EE + k0, EE, lane);
            acc[ns] = wmma16(a, b, acc[ns]);
        }
    }
    #pragma unroll
    for (int ns = 0; ns < 4; ++ns) {
        int f = ng * 64 + ns * 16 + n;
        float bi = bo[f];
        #pragma unroll
        for (int v = 0; v < 8; ++v) {
            int row = mt * 16 + v + 8 * half;
            out[(size_t)row * EE + f] = acc[ns][v] + bi;
        }
    }
}

// ---------------- launch ----------------
extern "C" void kernel_launch(void* const* d_in, const int* in_sizes, int n_in,
                              void* d_out, int out_size, void* d_ws, size_t ws_size,
                              hipStream_t stream) {
    const float* x  = (const float*)d_in[0];
    const float* Wq = (const float*)d_in[1];
    const float* bq = (const float*)d_in[2];
    const float* Wk = (const float*)d_in[3];
    const float* bk = (const float*)d_in[4];
    const float* Wv = (const float*)d_in[5];
    const float* bv = (const float*)d_in[6];
    const float* Wo = (const float*)d_in[7];
    const float* bo = (const float*)d_in[8];

    float* out  = (float*)d_out;
    float* attn = out + (size_t)BB * SS * EE;

    // workspace layout (≈72 MB): xf16 (aliased by ctx) | 4 weights f16 | Q | K | Vt
    char* w = (char*)d_ws;
    _Float16* xh  = (_Float16*)w;                              // 16 MB
    _Float16* wqh = (_Float16*)(w + ((size_t)16 << 20));       // 2 MB each
    _Float16* wkh = wqh + (size_t)EE * EE;
    _Float16* wvh = wkh + (size_t)EE * EE;
    _Float16* woh = wvh + (size_t)EE * EE;
    _Float16* Qw  = (_Float16*)(w + ((size_t)24 << 20));       // 16 MB
    _Float16* Kw  = (_Float16*)(w + ((size_t)40 << 20));       // 16 MB
    _Float16* Vtw = (_Float16*)(w + ((size_t)56 << 20));       // 16 MB
    _Float16* ctx = xh;   // x f16 is dead after qkv_proj; reuse for ctx

    // 1. fp32 -> fp16 conversions
    cvt_f32_f16<<<4096, 256, 0, stream>>>(x,  xh,  BB * SS * EE);
    cvt_f32_f16<<<1024, 256, 0, stream>>>(Wq, wqh, EE * EE);
    cvt_f32_f16<<<1024, 256, 0, stream>>>(Wk, wkh, EE * EE);
    cvt_f32_f16<<<1024, 256, 0, stream>>>(Wv, wvh, EE * EE);
    cvt_f32_f16<<<1024, 256, 0, stream>>>(Wo, woh, EE * EE);

    // 2. QKV projections: 3 * 512 * 16 wave-tiles (16x64 each), 8 waves/block
    qkv_proj<<<(3 * 512 * 16) / 8, 256, 0, stream>>>(xh, wqh, wkh, wvh, bq, bk, bv, Qw, Kw, Vtw);

    // 3. zero attn_avg region (accumulated via atomics)
    zero_f32<<<8192, 256, 0, stream>>>(attn, (size_t)BB * SS * SS);

    // 4. attention: B*H*(S/16) = 8192 waves, 4 waves/block
    attn_kernel<<<8192 / 4, 128, 0, stream>>>(Qw, Kw, Vtw, ctx, attn);

    // 5. output projection: 512*16 wave-tiles (16x64 each), 8 waves/block
    out_proj<<<(512 * 16) / 8, 256, 0, stream>>>(ctx, woh, bo, out);
}
